// BlockedMLP_32091995635878
// MI455X (gfx1250) — matrix operand order repaired and checked by reading, not compiled, not run on hardware
//
#include <hip/hip_runtime.h>
#include <hip/hip_bf16.h>

// ---------------------------------------------------------------------------
// BlockedMLP for MI455X (gfx1250):
//   h1 = relu(x @ w1^T + b1), h2 = relu(BSR(h1)), out = h2 @ w3^T + b3
// bf16 WMMA (f32 accum) everywhere. Tile staging:
//   - dense GEMMs: Tensor Data Mover (tensor_load_to_lds) with hardware LDS
//     row padding, double buffered, s_wait_tensorcnt pipeline
//   - BSR layer:   global_load_async_to_lds_b128 + s_wait_asynccnt pipeline
// ---------------------------------------------------------------------------

typedef __bf16 bf16;
typedef __attribute__((ext_vector_type(16))) __bf16 bf16x16;
typedef __attribute__((ext_vector_type(8)))  __bf16 bf16x8;
typedef __attribute__((ext_vector_type(8)))  float  f32x8;
typedef __attribute__((ext_vector_type(4)))  float  f32x4;
typedef __attribute__((ext_vector_type(4)))  unsigned int v4u;
typedef __attribute__((ext_vector_type(8)))  int v8i;
typedef __attribute__((ext_vector_type(4)))  int v4i;

__device__ __forceinline__ unsigned lds_addr_of(const void* p) {
  // flat shared-aperture address: low 32 bits are the LDS byte offset
  return (unsigned)(uintptr_t)p;
}

// CDNA5 async copy: global memory -> LDS, 16B per lane, tracked by ASYNCcnt.
__device__ __forceinline__ void async_copy_b128(unsigned lds_addr, const void* gaddr) {
  asm volatile("global_load_async_to_lds_b128 %0, %1, off"
               :: "v"(lds_addr), "v"(gaddr) : "memory");
}

// CDNA5 TDM: one descriptor moves a whole 2D tile global->LDS with hardware
// LDS row padding. D# packing per CDNA5 ISA.
//  - data_size code 1 (2-byte elements)
//  - tile_elems_x elements per row, tile_rows rows
//  - tensor row stride = stride_elems (elements)
//  - pad: every 2^(pad_i+1) DWORDs stored, insert (pad_a+1) DWORDs of padding
__device__ __forceinline__ void tdm_load_2d(unsigned lds_addr, const void* gptr,
                                            unsigned tile_elems_x, unsigned tile_rows,
                                            unsigned long stride_elems,
                                            unsigned pad_i, unsigned pad_a) {
  unsigned long ga = (unsigned long)gptr;  // 57-bit global byte address
  v4u g0;
  g0[0] = 1u;                                           // count=1, user descriptor
  g0[1] = lds_addr;                                     // LDS byte address
  g0[2] = (unsigned)(ga & 0xffffffffu);                 // global_addr[31:0]
  g0[3] = (unsigned)((ga >> 32) & 0x01ffffffu) | (2u << 30);  // addr[56:32], type=2
  const unsigned td0 = 0x7fffffffu, td1 = 0x7fffffffu;  // tensor dims (OOB bound only)
  v8i g1;
  g1[0] = (int)((1u << 16)            // data_size = 2 bytes
              | (1u << 20)            // pad_enable
              | (pad_i << 22)         // pad_interval
              | (pad_a << 25));       // pad_amount
  g1[1] = (int)((td0 & 0xffffu) << 16);                    // atomic_addr=0 | td0[15:0]
  g1[2] = (int)((td0 >> 16) | ((td1 & 0xffffu) << 16));    // td0[31:16] | td1[15:0]
  g1[3] = (int)((td1 >> 16) | (tile_elems_x << 16));       // td1[31:16] | tile_dim0
  g1[4] = (int)(tile_rows & 0xffffu);                      // tile_dim1 | tile_dim2=0
  g1[5] = (int)(stride_elems & 0xffffffffu);               // dim0_stride[31:0]
  g1[6] = (int)((stride_elems >> 32) & 0xffffu);           // dim0_stride[47:32]
  g1[7] = 0;                                               // dim1_stride (unused, 2D)
  v4i z4 = {};                                             // groups 2/3: zero (2D)
  v8i z8 = {};
  __builtin_amdgcn_tensor_load_to_lds(g0, g1, z4, z4, z8, 0);
}

// ---- f32 -> bf16 conversion -------------------------------------------------
__global__ __launch_bounds__(256) void f32_to_bf16_kernel(
    const float* __restrict__ in, bf16* __restrict__ out, int n) {
  int i = (blockIdx.x * blockDim.x + threadIdx.x) * 4;
  if (i + 3 < n) {
    f32x4 v = *(const f32x4*)(in + i);
#pragma unroll
    for (int k = 0; k < 4; ++k) out[i + k] = (bf16)v[k];
  } else {
    for (int k = i; k < n; ++k) out[k] = (bf16)in[k];
  }
}

// ---- WMMA fragment loads from LDS ------------------------------------------
// A 16x32 (MxK), 16-bit: lane half 0 -> K = {0..7, 16..23}, half 1 -> {8..15, 24..31}
__device__ __forceinline__ bf16x16 ldsFragA(const bf16* base, int stride, int lane) {
  const int half = (lane >> 4) & 1;
  const int r = lane & 15;
  const bf16* p = base + r * stride + half * 8;
  bf16x8 lo = *(const bf16x8*)(p);
  bf16x8 hi = *(const bf16x8*)(p + 16);
  bf16x16 f;
#pragma unroll
  for (int i = 0; i < 8; ++i) { f[i] = lo[i]; f[i + 8] = hi[i]; }
  return f;
}

// B 32x16 (KxN), 16-bit: lane half 0 -> K = {0..7, 16..23}, half 1 -> {8..15, 24..31} (N=lane&15)
__device__ __forceinline__ bf16x16 ldsFragB(const bf16* base, int stride, int lane) {
  const int half = (lane >> 4) & 1;
  const int r = lane & 15;  // N index
  const bf16* p = base + r * stride + half * 8;
  bf16x8 lo = *(const bf16x8*)(p);
  bf16x8 hi = *(const bf16x8*)(p + 16);
  bf16x16 f;
#pragma unroll
  for (int i = 0; i < 8; ++i) { f[i] = lo[i]; f[i + 8] = hi[i]; }
  return f;
}

// ---- dense GEMM: C[M,N] = act(A[M,K] @ W[N,K]^T + bias) --------------------
// 256 threads = 8 waves; WG tile 128x128; wave tile 64x32 (4x2 accumulators).
// Double-buffered LDS; tiles staged by TDM (wave 0 issues 2 descriptors/step).
// 128x32-elem tile, row = 64B = 16 DWORDs -> pad_interval code 3 (16 DWORDs),
// pad_amount code 3 (4 DWORDs = 16B) => LDS row stride 80B = LDT 40 elements.
template <bool RELU, typename OutT>
__global__ __launch_bounds__(256) void gemm_bias_kernel(
    const bf16* __restrict__ A, const bf16* __restrict__ W,
    const float* __restrict__ bias, OutT* __restrict__ C,
    int M, int N, int K) {
  constexpr int LDT = 40;  // 32 + 8 pad (80B rows, 16B aligned)
  __shared__ __align__(16) bf16 As[2][128 * LDT];
  __shared__ __align__(16) bf16 Bs[2][128 * LDT];

  const int tid = threadIdx.x;
  const int lane = tid & 31;
  const int wid = tid >> 5;
  const int wr = wid >> 2;  // 0..1  -> 64 rows each
  const int wc = wid & 3;   // 0..3  -> 32 cols each
  const int m0 = blockIdx.y * 128;
  const int n0 = blockIdx.x * 128;
  const int nk = K >> 5;

  f32x8 acc[4][2];
#pragma unroll
  for (int i = 0; i < 4; ++i)
#pragma unroll
    for (int j = 0; j < 2; ++j) acc[i][j] = {};

  // wave 0 stages one 128x32 A tile + 128x32 B tile into buffer `buf` via TDM
  auto stage = [&](int k0, int buf) {
    if (tid < 32) {
      tdm_load_2d(lds_addr_of(&As[buf][0]), &A[(size_t)m0 * K + k0], 32, 128,
                  (unsigned long)K, 3, 3);
      tdm_load_2d(lds_addr_of(&Bs[buf][0]), &W[(size_t)n0 * K + k0], 32, 128,
                  (unsigned long)K, 3, 3);
    }
  };

  stage(0, 0);
  for (int it = 0; it < nk; ++it) {
    const int buf = it & 1;
    if (it + 1 < nk) {
      stage((it + 1) << 5, buf ^ 1);
      // TENSORcnt is in-order: current pair done when only next pair remains
      if (tid < 32) __builtin_amdgcn_s_wait_tensorcnt(2);
    } else {
      if (tid < 32) __builtin_amdgcn_s_wait_tensorcnt(0);
    }
    __syncthreads();  // propagate wave-0 completion to all waves

    bf16x16 bfrag[2];
#pragma unroll
    for (int j = 0; j < 2; ++j)
      bfrag[j] = ldsFragB(&Bs[buf][(wc * 32 + j * 16) * LDT], LDT, lane);
#pragma unroll
    for (int i = 0; i < 4; ++i) {
      bf16x16 afrag = ldsFragA(&As[buf][(wr * 64 + i * 16) * LDT], LDT, lane);
#pragma unroll
      for (int j = 0; j < 2; ++j)
        acc[i][j] = __builtin_amdgcn_wmma_f32_16x16x32_bf16(
            false, afrag, false, bfrag[j], (short)0, acc[i][j], false, false);
    }
    __syncthreads();  // all waves done reading buf before it is restaged
  }

  // C/D layout: lane half h, VGPR v -> M = 8*h + v, N = lane&15
  const int half = lane >> 4;
  const int nl = lane & 15;
#pragma unroll
  for (int i = 0; i < 4; ++i) {
#pragma unroll
    for (int j = 0; j < 2; ++j) {
      int n = n0 + wc * 32 + j * 16 + nl;
      float bv = bias ? bias[n] : 0.0f;
#pragma unroll
      for (int v = 0; v < 8; ++v) {
        int m = m0 + wr * 64 + i * 16 + half * 8 + v;
        float val = acc[i][j][v] + bv;
        if (RELU) val = val > 0.0f ? val : 0.0f;
        C[(size_t)m * N + n] = (OutT)val;
      }
    }
  }
}

// ---- BSR layer: h2[:, br*64:+64] = relu(sum_j h1[:, col[j]*64:+64] @ V[j]^T)
// WG tile: 128 rows x 64 cols (one block-row). 8 waves as 4x2; wave 32x32.
// Double-buffered over the nnz block list (async-to-LDS, 6 ops/thread/block).
__global__ __launch_bounds__(256) void bsr_relu_kernel(
    const bf16* __restrict__ A,     // h1 [M, H] bf16
    const bf16* __restrict__ vals,  // [nnzb, 64, 64] bf16 (out-major, in-minor)
    const int* __restrict__ crow, const int* __restrict__ colid,
    bf16* __restrict__ C,           // h2 [M, H]
    int M, int Hdim) {
  constexpr int LDT = 72;  // 64 + 8 pad (144B rows)
  __shared__ __align__(16) bf16 As[2][128 * LDT];
  __shared__ __align__(16) bf16 Bs[2][64 * LDT];

  const int tid = threadIdx.x;
  const int lane = tid & 31;
  const int wid = tid >> 5;
  const int wr = wid >> 1;  // 0..3 -> 32 rows each
  const int wc = wid & 1;   // 0..1 -> 32 cols each
  const int m0 = blockIdx.x * 128;
  const int brow = blockIdx.y;

  const int js = crow[brow];
  const int je = crow[brow + 1];
  const int nj = je - js;

  f32x8 acc[2][2];
#pragma unroll
  for (int i = 0; i < 2; ++i)
#pragma unroll
    for (int j = 0; j < 2; ++j) acc[i][j] = {};

  auto stage = [&](int j, int buf) {
    const int bc = colid[js + j];
#pragma unroll
    for (int s = 0; s < 4; ++s) {  // A: 1024 chunks / 256 threads
      int idx = tid + s * 256;
      int r = idx >> 3, c = (idx & 7) << 3;
      async_copy_b128(lds_addr_of(&As[buf][r * LDT + c]),
                      &A[(size_t)(m0 + r) * Hdim + bc * 64 + c]);
    }
    const bf16* vb = vals + (size_t)(js + j) * 64 * 64;
#pragma unroll
    for (int s = 0; s < 2; ++s) {  // B: 512 chunks / 256 threads
      int idx = tid + s * 256;
      int r = idx >> 3, c = (idx & 7) << 3;
      async_copy_b128(lds_addr_of(&Bs[buf][r * LDT + c]), &vb[r * 64 + c]);
    }
  };

  if (nj > 0) stage(0, 0);
  for (int j = 0; j < nj; ++j) {
    const int buf = j & 1;
    if (j + 1 < nj) {
      stage(j + 1, buf ^ 1);
      asm volatile("s_wait_asynccnt 0x6" ::: "memory");
    } else {
      asm volatile("s_wait_asynccnt 0x0" ::: "memory");
    }
    __syncthreads();

#pragma unroll
    for (int ks = 0; ks < 64; ks += 32) {
      bf16x16 bfrag[2];
#pragma unroll
      for (int jj = 0; jj < 2; ++jj)
        bfrag[jj] = ldsFragB(&Bs[buf][(wc * 32 + jj * 16) * LDT + ks], LDT, lane);
#pragma unroll
      for (int i = 0; i < 2; ++i) {
        bf16x16 afrag = ldsFragA(&As[buf][(wr * 32 + i * 16) * LDT + ks], LDT, lane);
#pragma unroll
        for (int jj = 0; jj < 2; ++jj)
          acc[i][jj] = __builtin_amdgcn_wmma_f32_16x16x32_bf16(
              false, afrag, false, bfrag[jj], (short)0, acc[i][jj], false, false);
      }
    }
    __syncthreads();
  }

  const int half = lane >> 4;
  const int nl = lane & 15;
#pragma unroll
  for (int i = 0; i < 2; ++i) {
#pragma unroll
    for (int jj = 0; jj < 2; ++jj) {
      int n = brow * 64 + wc * 32 + jj * 16 + nl;
#pragma unroll
      for (int v = 0; v < 8; ++v) {
        int m = m0 + wr * 32 + i * 16 + half * 8 + v;
        float val = acc[i][jj][v];
        val = val > 0.0f ? val : 0.0f;
        C[(size_t)m * Hdim + n] = (bf16)val;
      }
    }
  }
}

// ---------------------------------------------------------------------------
extern "C" void kernel_launch(void* const* d_in, const int* in_sizes, int n_in,
                              void* d_out, int out_size, void* d_ws, size_t ws_size,
                              hipStream_t stream) {
  (void)n_in; (void)out_size; (void)ws_size;
  const int B = 4096, D_IN = 1024, H = 4096, D_OUT = 1024;

  const float* x      = (const float*)d_in[0];
  const float* w1     = (const float*)d_in[1];
  const float* b1     = (const float*)d_in[2];
  const float* values = (const float*)d_in[3];
  const float* w3     = (const float*)d_in[4];
  const float* b3     = (const float*)d_in[5];
  const int*   crow   = (const int*)d_in[6];
  const int*   col    = (const int*)d_in[7];
  const int nnz_elems = in_sizes[3];  // nnzb * 64 * 64

  char* ws = (char*)d_ws;
  size_t off = 0;
  auto take = [&](size_t bytes) -> char* {
    char* p = ws + off;
    off = (off + bytes + 255) & ~(size_t)255;
    return p;
  };
  bf16* xb  = (bf16*)take((size_t)B * D_IN * 2);
  bf16* w1b = (bf16*)take((size_t)H * D_IN * 2);
  bf16* w3b = (bf16*)take((size_t)D_OUT * H * 2);
  bf16* vb  = (bf16*)take((size_t)nnz_elems * 2);
  bf16* h1  = (bf16*)take((size_t)B * H * 2);
  bf16* h2  = (bf16*)take((size_t)B * H * 2);

  auto conv = [&](const float* src, bf16* dst, int n) {
    int blocks = (n + 256 * 4 - 1) / (256 * 4);
    f32_to_bf16_kernel<<<blocks, 256, 0, stream>>>(src, dst, n);
  };
  conv(x, xb, B * D_IN);
  conv(w1, w1b, H * D_IN);
  conv(w3, w3b, D_OUT * H);
  conv(values, vb, nnz_elems);

  // Layer 1: h1 = relu(x @ w1^T + b1)
  gemm_bias_kernel<true, bf16>
      <<<dim3(H / 128, B / 128), 256, 0, stream>>>(xb, w1b, b1, h1, B, H, D_IN);

  // Layer 2: h2 = relu(BSR(h1))
  bsr_relu_kernel<<<dim3(B / 128, H / 64), 256, 0, stream>>>(h1, vb, crow, col, h2, B, H);

  // Layer 3: out = h2 @ w3^T + b3 (f32 output)
  gemm_bias_kernel<false, float>
      <<<dim3(D_OUT / 128, B / 128), 256, 0, stream>>>(h2, w3b, b3, (float*)d_out, B, D_OUT, H);
}
